// TPAVIModule_13365938225874
// MI455X (gfx1250) — matrix operand order repaired and checked by reading, not compiled, not run on hardware
//
#include <hip/hip_runtime.h>

#define NB   4
#define CC   256
#define CI   128
#define TT   5
#define HW   784
#define LL   3920      // T*H*W
#define TILES 245      // LL/16, each tile stays within one frame (784 = 49*16)
#define WPB  4         // waves per block in main kernel
#define BPN  62        // ceil(245/4) blocks per batch image
#define CPAD 260       // LDS row stride (floats): 16B aligned, bank-spread

typedef __attribute__((ext_vector_type(16))) __bf16 v16bf;
typedef __attribute__((ext_vector_type(8)))  float  v8f;

__device__ __forceinline__ float wave_reduce_add(float v) {
#pragma unroll
  for (int off = 16; off > 0; off >>= 1) v += __shfl_xor(v, off, 32);
  return v;
}

// ---------------------------------------------------------------------------
// Kernel 1: Xsum(n,c,t) = sum_{h,w} x(n,c,t,h,w). One wave per (n,c,t).
// ---------------------------------------------------------------------------
__global__ void xsum_kernel(const float* __restrict__ x, float* __restrict__ Xsum) {
  int wave = blockIdx.x * 8 + (threadIdx.x >> 5);  // 0..5119 == (n*C+c)*T+t
  int lane = threadIdx.x & 31;
  const float4* base = (const float4*)(x + (size_t)wave * HW);
  float s = 0.f;
#pragma unroll
  for (int i = 0; i < 7; i++) {
    int idx = lane + 32 * i;
    if (idx < 196) { float4 v = base[idx]; s += v.x + v.y + v.z + v.w; }
  }
  s = wave_reduce_add(s);
  if (lane == 0) Xsum[wave] = s;
}

// ---------------------------------------------------------------------------
// Kernel 2: per-n precompute (tiny). audio_temp, phi, M (bf16, padded to 16
// rows), Gsum, folded WGs table and per-channel bias. One block per n.
// ---------------------------------------------------------------------------
__global__ void prep_kernel(const float* __restrict__ audio,
                            const float* __restrict__ alignW, const float* __restrict__ alignb,
                            const float* __restrict__ gW,     const float* __restrict__ gb,
                            const float* __restrict__ thetaW, const float* __restrict__ thetab,
                            const float* __restrict__ phiW,   const float* __restrict__ phib,
                            const float* __restrict__ WzW,    const float* __restrict__ Wzb,
                            const float* __restrict__ bng,    const float* __restrict__ bnb,
                            const float* __restrict__ bnm,    const float* __restrict__ bnv,
                            const float* __restrict__ Xsum,
                            __bf16* __restrict__ Mb, float* __restrict__ WGs,
                            float* __restrict__ Bb, float* __restrict__ out_at) {
  __shared__ float sAudio[TT * 128];
  __shared__ float sAT[TT * CC];
  __shared__ float sPhi[TT * CI];
  __shared__ float sG[TT * CI];
  __shared__ float sD[TT];

  int n = blockIdx.x, tid = threadIdx.x;
  for (int i = tid; i < TT * 128; i += 256) sAudio[i] = audio[n * TT * 128 + i];
  __syncthreads();

  { // audio_temp = audio @ alignW^T + alignb  (written to d_out tail too)
    int c = tid;
    for (int t = 0; t < TT; t++) {
      float acc = alignb[c];
      for (int ca = 0; ca < 128; ca++) acc += sAudio[t * 128 + ca] * alignW[c * 128 + ca];
      sAT[t * CC + c] = acc;
      out_at[(n * TT + t) * CC + c] = acc;
    }
  }
  __syncthreads();

  if (tid < CI) { // phi(t,o) and Gsum(t,o)
    int o = tid;
    for (int t = 0; t < TT; t++) {
      float acc = phib[o];
      for (int c = 0; c < CC; c++) acc += sAT[t * CC + c] * phiW[o * CC + c];
      sPhi[t * CI + o] = acc;
    }
    for (int t = 0; t < TT; t++) {
      float acc = (float)HW * gb[o];
      for (int c = 0; c < CC; c++) acc += gW[o * CC + c] * Xsum[(n * CC + c) * TT + t];
      sG[t * CI + o] = acc;
    }
  }
  __syncthreads();

  if (tid < TT) { // d(t) = thetab . phi(t,:)
    float acc = 0.f;
    for (int o = 0; o < CI; o++) acc += thetab[o] * sPhi[tid * CI + o];
    sD[tid] = acc;
  }
  { // M(t,c) = sum_o phi(t,o)*thetaW(o,c), padded to 16 rows, stored bf16
    int c = tid;
    for (int t = 0; t < 16; t++) {
      float m = 0.f;
      if (t < TT) for (int o = 0; o < CI; o++) m += sPhi[t * CI + o] * thetaW[o * CC + c];
      Mb[(n * 16 + t) * CC + c] = (__bf16)m;
    }
  }
  __syncthreads();

  { // fold Wz, 1/L, BN into WGs(t,c) and bias B(n,c)
    int c = tid;
    float inv  = bng[c] * rsqrtf(bnv[c] + 1e-5f);
    float bias = inv * (Wzb[c] - bnm[c]) + bnb[c];
    for (int t = 0; t < TT; t++) {
      float wg = 0.f;
      for (int o = 0; o < CI; o++) wg += WzW[c * CI + o] * sG[t * CI + o];
      float wgs = wg * inv * (1.0f / (float)LL);
      WGs[(n * TT + t) * CC + c] = wgs;
      bias += sD[t] * wgs;
    }
    Bb[n * CC + c] = bias;
  }
}

// ---------------------------------------------------------------------------
// Kernel 3: fused main pass. Per wave: one 16-token tile.
//   fT0 = X_tile(16x256) @ M^T  via 8x v_wmma_f32_16x16x32_bf16, then
//   z = LN( x + B + sum_t fT0(t)*WGs(t,:) ).
// ---------------------------------------------------------------------------
__global__ void __launch_bounds__(128) tpavi_main(const float* __restrict__ x,
                                                  const __bf16* __restrict__ Mb,
                                                  const float* __restrict__ WGs,
                                                  const float* __restrict__ Bb,
                                                  const float* __restrict__ lng,
                                                  const float* __restrict__ lnb,
                                                  float* __restrict__ out) {
  __shared__ __bf16 sM[16 * CC];
  __shared__ float  sWG[TT * CC];
  __shared__ float  sBias[CC];
  __shared__ float  sLg[CC];
  __shared__ float  sLb[CC];
  __shared__ float  sX[WPB][16 * CPAD];
  __shared__ float  sF[WPB][16 * 16];

  int n = blockIdx.x / BPN;
  int bi = blockIdx.x % BPN;
  int tid = threadIdx.x;
  int w = tid >> 5, lane = tid & 31;

  for (int i = tid; i < 16 * CC; i += 128) sM[i] = Mb[n * 16 * CC + i];
  for (int i = tid; i < TT * CC; i += 128) sWG[i] = WGs[n * TT * CC + i];
  for (int i = tid; i < CC; i += 128) {
    sBias[i] = Bb[n * CC + i]; sLg[i] = lng[i]; sLb[i] = lnb[i];
  }
  __syncthreads();

  // B fragments (invariant over tiles): lane = column t' (0..15), K range
  // selected by lane[4]; VGPR j holds K = base+2j, base+2j+1.
  v16bf Bf[8];
  {
    int col = lane & 15;
    int kb  = (lane & 16) ? 16 : 0;
#pragma unroll
    for (int kk = 0; kk < 8; kk++) {
      const __bf16* mr = &sM[col * CC + kk * 32 + kb];
      v16bf b;
#pragma unroll
      for (int j = 0; j < 16; j++) b[j] = mr[j];
      Bf[kk] = b;
    }
  }

  int tile = bi * WPB + w;
  if (tile >= TILES) return;           // whole-wave guard: EXEC all-ones at WMMA
  int l0 = tile * 16;
  float* sXw = sX[w];
  float* sFw = sF[w];
  const float* xn = x + (size_t)n * CC * LL;
  float* on = out + (size_t)n * CC * LL;

  // Phase A: coalesced b128 loads, token-major staging in LDS
  {
    int csub = lane >> 2, tq = lane & 3;
#pragma unroll 4
    for (int cb = 0; cb < 32; cb++) {
      int c = cb * 8 + csub;
      const float4 v = *(const float4*)(xn + (size_t)c * LL + l0 + tq * 4);
      sXw[(tq * 4 + 0) * CPAD + c] = v.x;
      sXw[(tq * 4 + 1) * CPAD + c] = v.y;
      sXw[(tq * 4 + 2) * CPAD + c] = v.z;
      sXw[(tq * 4 + 3) * CPAD + c] = v.w;
    }
  }
  __asm__ volatile("s_wait_dscnt 0" ::: "memory");

  // Phase B: 8x WMMA bf16, K=256. A uses contiguous K (k = 16*h + j), same as B.
  v8f acc = {0.f, 0.f, 0.f, 0.f, 0.f, 0.f, 0.f, 0.f};
  {
    int tok = lane & 15;
    int kb  = (lane & 16) ? 16 : 0;  // lanes16-31 hold K in {16..31}
#pragma unroll
    for (int kk = 0; kk < 8; kk++) {
      const float* xr = &sXw[tok * CPAD + kk * 32 + kb];
      v16bf a;
#pragma unroll
      for (int j = 0; j < 16; j++) a[j] = (__bf16)xr[j];
      acc = __builtin_amdgcn_wmma_f32_16x16x32_bf16(false, a, false, Bf[kk],
                                                    (short)0, acc, false, false);
    }
  }
  // Spill fT0 (D layout: VGPR r -> row r / r+8; col = lane&15)
  {
    int frame = lane & 15;
    int mo = (lane & 16) ? 8 : 0;
#pragma unroll
    for (int r = 0; r < 8; r++) sFw[(r + mo) * 16 + frame] = acc[r];
  }
  __asm__ volatile("s_wait_dscnt 0" ::: "memory");

  // Phase C: rank-5 channel update + folded BN bias + residual + LayerNorm
  for (int tok = 0; tok < 16; tok++) {
    float f0 = sFw[tok * 16 + 0], f1 = sFw[tok * 16 + 1], f2 = sFw[tok * 16 + 2];
    float f3 = sFw[tok * 16 + 3], f4 = sFw[tok * 16 + 4];
    float zp[8]; float s = 0.f, sq = 0.f;
#pragma unroll
    for (int i = 0; i < 8; i++) {
      int c = lane + 32 * i;
      float z = sXw[tok * CPAD + c] + sBias[c]
              + f0 * sWG[0 * CC + c] + f1 * sWG[1 * CC + c] + f2 * sWG[2 * CC + c]
              + f3 * sWG[3 * CC + c] + f4 * sWG[4 * CC + c];
      zp[i] = z; s += z; sq += z * z;
    }
    s = wave_reduce_add(s); sq = wave_reduce_add(sq);
    float mu  = s * (1.f / CC);
    float var = sq * (1.f / CC) - mu * mu;
    float rs  = rsqrtf(var + 1e-5f);
#pragma unroll
    for (int i = 0; i < 8; i++) {
      int c = lane + 32 * i;
      sXw[tok * CPAD + c] = (zp[i] - mu) * rs * sLg[c] + sLb[c];
    }
  }
  __asm__ volatile("s_wait_dscnt 0" ::: "memory");

  // Phase D: coalesced b128 store
  {
    int csub = lane >> 2, tq = lane & 3;
#pragma unroll 4
    for (int cb = 0; cb < 32; cb++) {
      int c = cb * 8 + csub;
      float4 v;
      v.x = sXw[(tq * 4 + 0) * CPAD + c];
      v.y = sXw[(tq * 4 + 1) * CPAD + c];
      v.z = sXw[(tq * 4 + 2) * CPAD + c];
      v.w = sXw[(tq * 4 + 3) * CPAD + c];
      *(float4*)(on + (size_t)c * LL + l0 + tq * 4) = v;
    }
  }
}

// ---------------------------------------------------------------------------
extern "C" void kernel_launch(void* const* d_in, const int* in_sizes, int n_in,
                              void* d_out, int out_size, void* d_ws, size_t ws_size,
                              hipStream_t stream) {
  const float* x      = (const float*)d_in[0];
  const float* audio  = (const float*)d_in[1];
  const float* alignW = (const float*)d_in[2];
  const float* alignb = (const float*)d_in[3];
  const float* gW     = (const float*)d_in[4];
  const float* gb     = (const float*)d_in[5];
  const float* thetaW = (const float*)d_in[6];
  const float* thetab = (const float*)d_in[7];
  const float* phiW   = (const float*)d_in[8];
  const float* phib   = (const float*)d_in[9];
  const float* WzW    = (const float*)d_in[10];
  const float* Wzb    = (const float*)d_in[11];
  const float* bng    = (const float*)d_in[12];
  const float* bnb    = (const float*)d_in[13];
  const float* bnm    = (const float*)d_in[14];
  const float* bnv    = (const float*)d_in[15];
  const float* lng    = (const float*)d_in[16];
  const float* lnb    = (const float*)d_in[17];

  float* out    = (float*)d_out;
  float* out_at = out + (size_t)NB * CC * LL;   // audio_temp tail (N,T,C)

  // workspace layout (needs ~78 KB)
  char*   ws   = (char*)d_ws;
  float*  Xsum = (float*)ws;                    // 5120 f32
  __bf16* Mb   = (__bf16*)(ws + 20480);         // N*16*256 bf16
  float*  WGsp = (float*)(ws + 53248);          // N*5*256 f32
  float*  Bb   = (float*)(ws + 73728);          // N*256 f32

  xsum_kernel<<<640, 256, 0, stream>>>(x, Xsum);
  prep_kernel<<<NB, 256, 0, stream>>>(audio, alignW, alignb, gW, gb, thetaW, thetab,
                                      phiW, phib, WzW, Wzb, bng, bnb, bnm, bnv,
                                      Xsum, Mb, WGsp, Bb, out_at);
  tpavi_main<<<NB * BPN, 128, 0, stream>>>(x, Mb, WGsp, Bb, lng, lnb, out);
}
